// CrossViewSelfAttentionFusion_56169582297653
// MI455X (gfx1250) — compile-verified, matrix operand order checked by reading, not run on hardware
//
#include <hip/hip_runtime.h>

#define SEQ 4096
#define CDIM 256
#define NHEAD 8
#define DHEAD 32
#define NLAYER 3
#define DFF 1024
#define KP 40  // padded LDS row (halves); 80 bytes, 16B aligned

typedef __attribute__((ext_vector_type(16))) _Float16 v16h;
typedef __attribute__((ext_vector_type(8)))  _Float16 v8h;
typedef __attribute__((ext_vector_type(8)))  float    v8f;
typedef __attribute__((ext_vector_type(4)))  int      v4i;

typedef __attribute__((address_space(1))) v4i glob_v4i;
typedef __attribute__((address_space(3))) v4i lds_v4i;

#if defined(__has_builtin)
#if __has_builtin(__builtin_amdgcn_global_load_async_to_lds_b128)
#define ASYNC_LDS 1
#endif
#endif

// stage one 16B chunk global -> LDS (async DMA when available)
__device__ inline void stage_chunk(const _Float16* src, _Float16* dst) {
#ifdef ASYNC_LDS
  __builtin_amdgcn_global_load_async_to_lds_b128(
      (glob_v4i*)src, (lds_v4i*)dst, 0, 0);
#else
  *(v8h*)dst = *(const v8h*)src;
#endif
}

__device__ inline void wait_async_le1() {
#if defined(__has_builtin) && __has_builtin(__builtin_amdgcn_s_wait_asynccnt)
  __builtin_amdgcn_s_wait_asynccnt(1);
#else
  asm volatile("s_wait_asynccnt 1" ::: "memory");
#endif
}
__device__ inline void wait_async_le0() {
#if defined(__has_builtin) && __has_builtin(__builtin_amdgcn_s_wait_asynccnt)
  __builtin_amdgcn_s_wait_asynccnt(0);
#else
  asm volatile("s_wait_asynccnt 0" ::: "memory");
#endif
}

// ---------------------------------------------------------------------------
// Fragment load for V_WMMA_F32_16X16X32_F16 operands (A or B), row-major
// source with leading dimension `ld` (elements). Lane l holds row (l&15);
// elements 0..7 are k = kh*8 + {0..7}, elements 8..15 are k = 16+kh*8+{0..7},
// kh = l>>4.  Two 16-byte loads per lane.
// ---------------------------------------------------------------------------
__device__ inline v16h load_frag(const _Float16* base, int ld, int lane) {
  int r  = lane & 15;
  int kh = (lane >> 4) & 1;
  const _Float16* p = base + (size_t)r * ld + kh * 8;
  v8h lo = *(const v8h*)(p);
  v8h hi = *(const v8h*)(p + 16);
  v16h f;
#pragma unroll
  for (int i = 0; i < 8; ++i) { f[i] = lo[i]; f[i + 8] = hi[i]; }
  return f;
}

__device__ inline v8f wmma16(v16h a, v16h b, v8f c) {
  return __builtin_amdgcn_wmma_f32_16x16x32_f16(false, a, false, b,
                                                (short)0, c, false, false);
}

__device__ inline float gelu_exact(float x) {
  return 0.5f * x * (1.f + erff(x * 0.70710678118654752f));
}

// ---------------------------------------------------------------------------
// GEMM: out[M,N] = act(A[M,K] @ W[N,K]^T + bias) (+ residual)
// Block = 8 waves -> tile 128M x 64N; each wave owns 16M x 64N (4 wmma accs).
// The 64-row W strip per 32-wide k-step is staged into a double-buffered LDS
// slab with async global->LDS DMA: the k+1 strip is issued before computing
// on the k strip, and we only drain to asynccnt<=1 so one DMA stays in
// flight behind the wmma work.  Requires M%128==0, N%64==0, K%32==0.
// grid = (N/64, M/128), block = 256.
// ---------------------------------------------------------------------------
__global__ __launch_bounds__(256)
void gemm_f16w(const _Float16* __restrict__ A, int lda,
               const _Float16* __restrict__ W, int ldw,
               const float* __restrict__ bias,
               float* __restrict__ outF, _Float16* __restrict__ outH,
               const float* __restrict__ residual,
               int M, int N, int K, int do_gelu) {
  __shared__ _Float16 Ws[2][64 * KP];
  int lane = threadIdx.x & 31;
  int wave = threadIdx.x >> 5;
  int m0 = blockIdx.y * 128 + wave * 16;
  int n0 = blockIdx.x * 64;

  // this thread's staging slot: one 16B chunk of the 64x32 W strip
  int srow = threadIdx.x >> 2;          // 0..63
  int sch  = (threadIdx.x & 3) * 8;     // 0,8,16,24
  const _Float16* wsrc = W + (size_t)(n0 + srow) * ldw + sch;
  _Float16* wdst[2] = {&Ws[0][srow * KP + sch], &Ws[1][srow * KP + sch]};
  const _Float16* abase = A + (size_t)m0 * lda;

  int nIter = K >> 5;
  stage_chunk(wsrc, wdst[0]);  // prologue: k-step 0 into buffer 0

  v8f acc0 = {}, acc1 = {}, acc2 = {}, acc3 = {};
  for (int it = 0; it < nIter; ++it) {
    int cur = it & 1;
    if (it + 1 < nIter) {
      stage_chunk(wsrc + (size_t)(it + 1) * 32, wdst[cur ^ 1]);  // prefetch
      wait_async_le1();   // current strip resident; prefetch stays in flight
    } else {
      wait_async_le0();
    }
    __syncthreads();

    const _Float16* buf = &Ws[cur][0];
    v16h a  = load_frag(abase + (size_t)it * 32, lda, lane);
    v16h b0 = load_frag(buf, KP, lane);
    v16h b1 = load_frag(buf + 16 * KP, KP, lane);
    v16h b2 = load_frag(buf + 32 * KP, KP, lane);
    v16h b3 = load_frag(buf + 48 * KP, KP, lane);
    acc0 = wmma16(a, b0, acc0);
    acc1 = wmma16(a, b1, acc1);
    acc2 = wmma16(a, b2, acc2);
    acc3 = wmma16(a, b3, acc3);
    __syncthreads();  // all waves done reading `buf` before it is re-staged
  }

  int mbase = m0 + ((lane >> 4) ? 8 : 0);
  v8f* accs[4] = {&acc0, &acc1, &acc2, &acc3};
#pragma unroll
  for (int j = 0; j < 4; ++j) {
    int n = n0 + j * 16 + (lane & 15);
    float bv = bias ? bias[n] : 0.f;
#pragma unroll
    for (int v = 0; v < 8; ++v) {
      int m = mbase + v;
      float val = (*accs[j])[v] + bv;
      if (do_gelu) val = gelu_exact(val);
      if (residual) val += residual[(size_t)m * N + n];
      if (outF) outF[(size_t)m * N + n] = val;
      if (outH) outH[(size_t)m * N + n] = (_Float16)val;
    }
  }
}

// ---------------------------------------------------------------------------
// Flash attention over the full 4096-token sequence, per head.
// qkv: [4096,768] f16 (q|k|v packed). grid = (SEQ/128, NHEAD), block = 256.
// Each wave owns a 16-query tile.  Computes S^T = K_blk @ Q^T so softmax is
// lane-local and the exp'd tile is directly the B fragment for O^T = V^T P^T.
// ---------------------------------------------------------------------------
__global__ __launch_bounds__(256)
void flash_attn(const _Float16* __restrict__ qkv, _Float16* __restrict__ outO) {
  __shared__ _Float16 Ks[32 * KP];   // [key][d]
  __shared__ _Float16 Vt[32 * KP];   // [d][key]
  int lane = threadIdx.x & 31;
  int wave = threadIdx.x >> 5;
  int head = blockIdx.y;
  int q0 = blockIdx.x * 128 + wave * 16;

  // B fragment of Q^T: lane = query column, elements = head-dim
  v16h bq = load_frag(qkv + (size_t)q0 * 768 + head * DHEAD, 768, lane);

  v8f acc0 = {}, acc1 = {};
  float mrow = -1e30f, lrow = 0.f;
  const float scale = 0.17677669529663687f;  // 1/sqrt(32)
  const v8f zf = {};

  for (int kb = 0; kb < SEQ; kb += 32) {
    // cooperative stage: K block as-is, V block transposed
    for (int i = threadIdx.x; i < 32 * 32; i += 256) {
      int key = i >> 5, d = i & 31;
      const _Float16* row = qkv + (size_t)(kb + key) * 768 + head * DHEAD;
      Ks[key * KP + d] = row[256 + d];
      Vt[d * KP + key] = row[512 + d];
    }
    __syncthreads();

    v16h ak0 = load_frag(Ks, KP, lane);            // keys 0..15
    v16h ak1 = load_frag(Ks + 16 * KP, KP, lane);  // keys 16..31
    v8f s0 = wmma16(ak0, bq, zf);                  // S^T tiles
    v8f s1 = wmma16(ak1, bq, zf);

    float mx = -1e30f;
#pragma unroll
    for (int i = 0; i < 8; ++i) mx = fmaxf(mx, fmaxf(s0[i], s1[i]));
    mx = fmaxf(mx, __shfl_xor(mx, 16));
    float mnew = fmaxf(mrow, mx * scale);
    float corr = __expf(mrow - mnew);
    mrow = mnew;

    v16h bp;
    float psum = 0.f;
#pragma unroll
    for (int i = 0; i < 8; ++i) {
      float p0 = __expf(s0[i] * scale - mnew);
      float p1 = __expf(s1[i] * scale - mnew);
      psum += p0 + p1;
      bp[i] = (_Float16)p0;       // keys kh*8 + i   (this lane half)
      bp[i + 8] = (_Float16)p1;   // keys 16+kh*8+i
    }
    psum += __shfl_xor(psum, 16);
    lrow = lrow * corr + psum;
#pragma unroll
    for (int i = 0; i < 8; ++i) { acc0[i] *= corr; acc1[i] *= corr; }

    v16h av0 = load_frag(Vt, KP, lane);            // V^T rows d=0..15
    v16h av1 = load_frag(Vt + 16 * KP, KP, lane);  // V^T rows d=16..31
    acc0 = wmma16(av0, bp, acc0);                  // O^T accumulate
    acc1 = wmma16(av1, bp, acc1);
    __syncthreads();
  }

  float inv = 1.f / lrow;
  int q = q0 + (lane & 15);
  int dbase = (lane >> 4) ? 8 : 0;
  _Float16* orow = outO + (size_t)q * CDIM + head * DHEAD;
#pragma unroll
  for (int v = 0; v < 8; ++v) {
    orow[dbase + v]      = (_Float16)(acc0[v] * inv);
    orow[16 + dbase + v] = (_Float16)(acc1[v] * inv);
  }
}

// ---------------------------------------------------------------------------
// LayerNorm (row=256), one wave per row, optional exact-GELU fusion.
// ---------------------------------------------------------------------------
__global__ __launch_bounds__(256)
void ln_kernel(const float* __restrict__ x, const float* __restrict__ g,
               const float* __restrict__ b, _Float16* __restrict__ outH,
               int rows, int do_gelu) {
  int lane = threadIdx.x & 31;
  int row = (int)((blockIdx.x * blockDim.x + threadIdx.x) >> 5);
  if (row >= rows) return;
  const float* xr = x + (size_t)row * CDIM;
  float v[8], s = 0.f, s2 = 0.f;
#pragma unroll
  for (int i = 0; i < 8; ++i) {
    v[i] = xr[lane * 8 + i];
    s += v[i]; s2 += v[i] * v[i];
  }
  for (int off = 16; off; off >>= 1) {
    s += __shfl_xor(s, off);
    s2 += __shfl_xor(s2, off);
  }
  float mean = s * (1.f / CDIM);
  float var = s2 * (1.f / CDIM) - mean * mean;
  float rstd = rsqrtf(var + 1e-5f);
#pragma unroll
  for (int i = 0; i < 8; ++i) {
    int c = lane * 8 + i;
    float y = (v[i] - mean) * rstd * g[c] + b[c];
    if (do_gelu) y = gelu_exact(y);
    outH[(size_t)row * CDIM + c] = (_Float16)y;
  }
}

// x = features + view_emb[n] + pos_emb[l]
__global__ void add_embed(const float* __restrict__ f, const float* __restrict__ ve,
                          const float* __restrict__ pe, float* __restrict__ x) {
  int idx = blockIdx.x * blockDim.x + threadIdx.x;  // over 4*1024*256
  int c = idx & 255;
  int l = (idx >> 8) & 1023;
  int n = idx >> 18;
  x[idx] = f[idx] + ve[n * CDIM + c] + pe[l * CDIM + c];
}

__global__ void cvt_f32_f16(const float* __restrict__ in, _Float16* __restrict__ out, int n) {
  int i = blockIdx.x * blockDim.x + threadIdx.x;
  if (i < n) out[i] = (_Float16)in[i];
}

// kvh[(l*4+n)] = h[(n*1024+l)]
__global__ void permute_kv(const _Float16* __restrict__ h, _Float16* __restrict__ kvh) {
  int idx = blockIdx.x * blockDim.x + threadIdx.x;  // 4096*256
  int c = idx & 255;
  int r = idx >> 8;
  int n = r >> 10, l = r & 1023;
  kvh[(size_t)(l * 4 + n) * CDIM + c] = h[idx];
}

// Cross-view attention: per (l, head) wave; lane = head-dim; 4 keys.
__global__ __launch_bounds__(256)
void view_attn(const _Float16* __restrict__ Q1, const _Float16* __restrict__ K1,
               const _Float16* __restrict__ V1, _Float16* __restrict__ out) {
  int lane = threadIdx.x & 31;
  int wid = (int)((blockIdx.x * blockDim.x + threadIdx.x) >> 5);
  int l = wid >> 3, head = wid & 7;
  int base = head * DHEAD + lane;
  float q = (float)Q1[(size_t)l * CDIM + base];
  float s[4];
#pragma unroll
  for (int n = 0; n < 4; ++n) {
    float t = q * (float)K1[(size_t)(l * 4 + n) * CDIM + base];
    for (int off = 16; off; off >>= 1) t += __shfl_xor(t, off);
    s[n] = t * 0.17677669529663687f;
  }
  float mx = fmaxf(fmaxf(s[0], s[1]), fmaxf(s[2], s[3]));
  float p[4], sum = 0.f;
#pragma unroll
  for (int n = 0; n < 4; ++n) { p[n] = __expf(s[n] - mx); sum += p[n]; }
  float inv = 1.f / sum, o = 0.f;
#pragma unroll
  for (int n = 0; n < 4; ++n)
    o += p[n] * inv * (float)V1[(size_t)(l * 4 + n) * CDIM + base];
  out[(size_t)l * CDIM + base] = (_Float16)o;
}

// out = fused + rw * mean_n(features)
__global__ void final_add(const float* __restrict__ fused, const float* __restrict__ f,
                          const float* __restrict__ rw, float* __restrict__ out) {
  int idx = blockIdx.x * blockDim.x + threadIdx.x;  // 1024*256
  float m = 0.25f * (f[idx] + f[idx + 262144] + f[idx + 2 * 262144] + f[idx + 3 * 262144]);
  out[idx] = fused[idx] + rw[0] * m;
}

// ---------------------------------------------------------------------------
extern "C" void kernel_launch(void* const* d_in, const int* in_sizes, int n_in,
                              void* d_out, int out_size, void* d_ws, size_t ws_size,
                              hipStream_t stream) {
  const float* features   = (const float*)d_in[0];
  const float* view_emb   = (const float*)d_in[1];
  const float* pos_emb    = (const float*)d_in[2];
  const float* ln1_g      = (const float*)d_in[3];
  const float* ln1_b      = (const float*)d_in[4];
  const float* attn_qkv_w = (const float*)d_in[5];
  const float* attn_qkv_b = (const float*)d_in[6];
  const float* attn_out_w = (const float*)d_in[7];
  const float* attn_out_b = (const float*)d_in[8];
  const float* ln2_g      = (const float*)d_in[9];
  const float* ln2_b      = (const float*)d_in[10];
  const float* ff1_w      = (const float*)d_in[11];
  const float* ff1_b      = (const float*)d_in[12];
  const float* ff2_w      = (const float*)d_in[13];
  const float* ff2_b      = (const float*)d_in[14];
  const float* fln_g      = (const float*)d_in[15];
  const float* fln_b      = (const float*)d_in[16];
  const float* va_qkv_w   = (const float*)d_in[17];
  const float* va_qkv_b   = (const float*)d_in[18];
  const float* va_out_w   = (const float*)d_in[19];
  const float* va_out_b   = (const float*)d_in[20];
  const float* gq         = (const float*)d_in[21];
  const float* op1_w      = (const float*)d_in[22];
  const float* op1_b      = (const float*)d_in[23];
  const float* op_ln_g    = (const float*)d_in[24];
  const float* op_ln_b    = (const float*)d_in[25];
  const float* op2_w      = (const float*)d_in[26];
  const float* op2_b      = (const float*)d_in[27];
  const float* res_w      = (const float*)d_in[28];

  char* ws = (char*)d_ws;
  size_t off = 0;
  auto alloc = [&](size_t bytes) -> char* {
    char* p = ws + off;
    off += (bytes + 255) & ~(size_t)255;
    return p;
  };

  float*     x       = (float*)alloc((size_t)SEQ * CDIM * 4);
  _Float16*  h16     = (_Float16*)alloc((size_t)SEQ * CDIM * 2);
  _Float16*  qkv16   = (_Float16*)alloc((size_t)SEQ * 768 * 2);
  _Float16*  attn16  = (_Float16*)alloc((size_t)SEQ * CDIM * 2);
  _Float16*  mid16   = (_Float16*)alloc((size_t)SEQ * DFF * 2);
  _Float16*  hfin16  = (_Float16*)alloc((size_t)SEQ * CDIM * 2);
  _Float16*  kvh16   = (_Float16*)alloc((size_t)SEQ * CDIM * 2);
  _Float16*  gq16    = (_Float16*)alloc((size_t)1024 * CDIM * 2);
  _Float16*  q1_16   = (_Float16*)alloc((size_t)1024 * CDIM * 2);
  _Float16*  k1_16   = (_Float16*)alloc((size_t)SEQ * CDIM * 2);
  _Float16*  v1_16   = (_Float16*)alloc((size_t)SEQ * CDIM * 2);
  _Float16*  agg16   = (_Float16*)alloc((size_t)1024 * CDIM * 2);
  _Float16*  agg2_16 = (_Float16*)alloc((size_t)1024 * CDIM * 2);
  float*     h1f     = (float*)alloc((size_t)1024 * CDIM * 4);
  _Float16*  h2_16   = (_Float16*)alloc((size_t)1024 * CDIM * 2);
  float*     fusedf  = (float*)alloc((size_t)1024 * CDIM * 4);
  _Float16*  qkvW16  = (_Float16*)alloc((size_t)NLAYER * 768 * CDIM * 2);
  _Float16*  outW16  = (_Float16*)alloc((size_t)NLAYER * CDIM * CDIM * 2);
  _Float16*  ff1W16  = (_Float16*)alloc((size_t)NLAYER * DFF * CDIM * 2);
  _Float16*  ff2W16  = (_Float16*)alloc((size_t)NLAYER * CDIM * DFF * 2);
  _Float16*  vaQW16  = (_Float16*)alloc((size_t)768 * CDIM * 2);
  _Float16*  vaOW16  = (_Float16*)alloc((size_t)CDIM * CDIM * 2);
  _Float16*  op1W16  = (_Float16*)alloc((size_t)CDIM * CDIM * 2);
  _Float16*  op2W16  = (_Float16*)alloc((size_t)CDIM * CDIM * 2);
  (void)ws_size; (void)n_in; (void)in_sizes; (void)out_size;

  auto conv = [&](const float* src, _Float16* dst, int n) {
    cvt_f32_f16<<<(n + 255) / 256, 256, 0, stream>>>(src, dst, n);
  };
  conv(attn_qkv_w, qkvW16, NLAYER * 768 * CDIM);
  conv(attn_out_w, outW16, NLAYER * CDIM * CDIM);
  conv(ff1_w, ff1W16, NLAYER * DFF * CDIM);
  conv(ff2_w, ff2W16, NLAYER * CDIM * DFF);
  conv(va_qkv_w, vaQW16, 768 * CDIM);
  conv(va_out_w, vaOW16, CDIM * CDIM);
  conv(op1_w, op1W16, CDIM * CDIM);
  conv(op2_w, op2W16, CDIM * CDIM);
  conv(gq, gq16, 1024 * CDIM);

  auto gemm = [&](const _Float16* A, int lda, const _Float16* W, int ldw,
                  const float* bias, float* outF, _Float16* outH,
                  const float* res, int M, int N, int K, int gelu) {
    dim3 g(N / 64, M / 128);
    gemm_f16w<<<g, 256, 0, stream>>>(A, lda, W, ldw, bias, outF, outH, res, M, N, K, gelu);
  };

  add_embed<<<(SEQ * CDIM) / 256, 256, 0, stream>>>(features, view_emb, pos_emb, x);

  for (int i = 0; i < NLAYER; ++i) {
    // pre-LN self-attention
    ln_kernel<<<SEQ / 8, 256, 0, stream>>>(x, ln1_g + i * CDIM, ln1_b + i * CDIM, h16, SEQ, 0);
    gemm(h16, CDIM, qkvW16 + (size_t)i * 768 * CDIM, CDIM, attn_qkv_b + i * 768,
         nullptr, qkv16, nullptr, SEQ, 768, CDIM, 0);
    flash_attn<<<dim3(SEQ / 128, NHEAD), 256, 0, stream>>>(qkv16, attn16);
    gemm(attn16, CDIM, outW16 + (size_t)i * CDIM * CDIM, CDIM, attn_out_b + i * CDIM,
         x, nullptr, x, SEQ, CDIM, CDIM, 0);
    // pre-LN FFN with exact GELU
    ln_kernel<<<SEQ / 8, 256, 0, stream>>>(x, ln2_g + i * CDIM, ln2_b + i * CDIM, h16, SEQ, 0);
    gemm(h16, CDIM, ff1W16 + (size_t)i * DFF * CDIM, CDIM, ff1_b + i * DFF,
         nullptr, mid16, nullptr, SEQ, DFF, CDIM, 1);
    gemm(mid16, DFF, ff2W16 + (size_t)i * CDIM * DFF, DFF, ff2_b + i * CDIM,
         x, nullptr, x, SEQ, CDIM, DFF, 0);
  }

  // final LN + cross-view attention
  ln_kernel<<<SEQ / 8, 256, 0, stream>>>(x, fln_g, fln_b, hfin16, SEQ, 0);
  permute_kv<<<(SEQ * CDIM) / 256, 256, 0, stream>>>(hfin16, kvh16);
  gemm(gq16, CDIM, vaQW16, CDIM, va_qkv_b, nullptr, q1_16, nullptr, 1024, CDIM, CDIM, 0);
  gemm(kvh16, CDIM, vaQW16 + 256 * CDIM, CDIM, va_qkv_b + 256, nullptr, k1_16, nullptr, SEQ, CDIM, CDIM, 0);
  gemm(kvh16, CDIM, vaQW16 + 512 * CDIM, CDIM, va_qkv_b + 512, nullptr, v1_16, nullptr, SEQ, CDIM, CDIM, 0);
  view_attn<<<1024, 256, 0, stream>>>(q1_16, k1_16, v1_16, agg16);
  gemm(agg16, CDIM, vaOW16, CDIM, va_out_b, nullptr, agg2_16, nullptr, 1024, CDIM, CDIM, 0);

  // output head: Linear -> LN -> GELU -> Linear, + weighted mean residual
  gemm(agg2_16, CDIM, op1W16, CDIM, op1_b, h1f, nullptr, nullptr, 1024, CDIM, CDIM, 0);
  ln_kernel<<<1024 / 8, 256, 0, stream>>>(h1f, op_ln_g, op_ln_b, h2_16, 1024, 1);
  gemm(h2_16, CDIM, op2W16, CDIM, op2_b, fusedf, nullptr, nullptr, 1024, CDIM, CDIM, 0);
  final_add<<<(1024 * CDIM) / 256, 256, 0, stream>>>(fusedf, features, res_w, (float*)d_out);
}
